// SelfAttention_34007551049902
// MI455X (gfx1250) — matrix operand order repaired and checked by reading, not compiled, not run on hardware
//
#include <hip/hip_runtime.h>
#include <hip/hip_bf16.h>

#define NB 32
#define NC 512
#define HW 1024
#define NHEADS 8
#define HDIM 64

typedef __attribute__((ext_vector_type(16))) __bf16 v16bf;
typedef __attribute__((ext_vector_type(8)))  __bf16 v8bf;
typedef __attribute__((ext_vector_type(8)))  float  v8f;
typedef int v4i_vs __attribute__((vector_size(16)));   // matches builtin's param type

static __device__ __forceinline__ unsigned short f2bf(float f) {
    unsigned int u = __float_as_uint(f);
    u += 0x7fffu + ((u >> 16) & 1u);   // round-to-nearest-even
    return (unsigned short)(u >> 16);
}

// Load a v16bf WMMA fragment from two 16-byte-aligned 8-element chunks.
static __device__ __forceinline__ v16bf ld16(const unsigned short* p0, const unsigned short* p1) {
    const v8bf a = *(const v8bf*)p0;
    const v8bf b = *(const v8bf*)p1;
    v16bf r;
#pragma unroll
    for (int i = 0; i < 8; ++i) { r[i] = a[i]; r[i + 8] = b[i]; }
    return r;
}

// ---------------------------------------------------------------------------
// 32-byte global -> LDS staging. Prefer CDNA5 async path
// (global_load_async_to_lds_b128, ASYNCcnt) when the toolchain exposes it.
// ---------------------------------------------------------------------------
#if defined(__has_builtin)
#if __has_builtin(__builtin_amdgcn_global_load_async_to_lds_b128) && \
    __has_builtin(__builtin_amdgcn_s_wait_asynccnt)
#define HAVE_ASYNC_LDS 1
#endif
#endif

#if HAVE_ASYNC_LDS
#define WAIT_ASYNC(n) __builtin_amdgcn_s_wait_asynccnt(n)
static __device__ __forceinline__ void stage32(const unsigned short* g, unsigned short* l) {
    __builtin_amdgcn_global_load_async_to_lds_b128(
        (__attribute__((address_space(1))) v4i_vs*)g,
        (__attribute__((address_space(3))) v4i_vs*)l, 0, 0);
    __builtin_amdgcn_global_load_async_to_lds_b128(
        (__attribute__((address_space(1))) v4i_vs*)(g + 8),
        (__attribute__((address_space(3))) v4i_vs*)(l + 8), 0, 0);
}
#else
#define WAIT_ASYNC(n) do {} while (0)
static __device__ __forceinline__ void stage32(const unsigned short* g, unsigned short* l) {
    uint4 t0 = ((const uint4*)g)[0];
    uint4 t1 = ((const uint4*)g)[1];
    ((uint4*)l)[0] = t0;
    ((uint4*)l)[1] = t1;
}
#endif

// ---------------------------------------------------------------------------
// GroupNorm(8, 512) -> bf16, output transposed to [b][pixel][channel]
// ---------------------------------------------------------------------------
__global__ __launch_bounds__(256) void groupnorm_kernel(
    const float* __restrict__ x, const float* __restrict__ scale,
    const float* __restrict__ bias, unsigned short* __restrict__ xn)
{
    const int g = blockIdx.x, b = blockIdx.y;
    const int tid = threadIdx.x;
    const float* xg = x + ((size_t)b * NC + g * 64) * HW;

    float s = 0.f, s2 = 0.f;
    for (int i = tid; i < 64 * HW; i += 256) {
        float v = xg[i];
        s += v; s2 += v * v;
    }
    __shared__ float red[512];
    red[tid] = s; red[tid + 256] = s2;
    __syncthreads();
    for (int off = 128; off > 0; off >>= 1) {
        if (tid < off) { red[tid] += red[tid + off]; red[tid + 256] += red[tid + 256 + off]; }
        __syncthreads();
    }
    const float mean = red[0] * (1.f / 65536.f);
    const float var  = red[256] * (1.f / 65536.f) - mean * mean;
    const float rstd = __frsqrt_rn(var + 1e-5f);

    for (int i = tid; i < 64 * HW; i += 256) {
        int cl = i >> 10, px = i & 1023;
        int ch = g * 64 + cl;
        float v = (xg[i] - mean) * rstd * scale[ch] + bias[ch];
        xn[((size_t)b * HW + px) * NC + ch] = f2bf(v);
    }
}

__global__ void f32_to_bf16_kernel(const float* __restrict__ in,
                                   unsigned short* __restrict__ out, int n)
{
    int i = blockIdx.x * 256 + threadIdx.x;
    if (i < n) out[i] = f2bf(in[i]);
}

// ---------------------------------------------------------------------------
// GEMM: C[b][m][n] = A[b][m][k] * W[n][k] + bias[n]
// 128x128 block tile, K=512 by 32, double-buffered async LDS staging.
// MODE 0: bf16 out [b][HW][N]    (QKV)
// MODE 1: f32 out  [b][N][HW] + residual (proj)
// ---------------------------------------------------------------------------
template <int MODE>
__global__ __launch_bounds__(256) void gemm_kernel(
    const unsigned short* __restrict__ A,   // [b][HW][512] bf16
    const unsigned short* __restrict__ W,   // [N][512] bf16
    const float* __restrict__ bias,
    const float* __restrict__ resid,        // [b][N][HW] (MODE 1)
    void* __restrict__ outp, int N)
{
    const int b = blockIdx.z;
    const int m0 = blockIdx.y * 128;
    const int n0 = blockIdx.x * 128;
    const int tid = threadIdx.x;
    const int lane = tid & 31, wave = tid >> 5;
    const int wm = wave & 1, wn = wave >> 1;

    __shared__ __align__(32) unsigned short At[2][128 * 32];
    __shared__ __align__(32) unsigned short Bt[2][128 * 32];

    v8f acc[4][2];
#pragma unroll
    for (int i = 0; i < 4; ++i)
#pragma unroll
        for (int j = 0; j < 2; ++j)
#pragma unroll
            for (int v = 0; v < 8; ++v) acc[i][j][v] = 0.f;

    const size_t arow = ((size_t)b * HW + m0) * 512;
    const int r = tid >> 1, half = tid & 1;
    const unsigned short* aSrc = A + arow + (size_t)r * 512 + half * 16;
    const unsigned short* bSrc = W + (size_t)(n0 + r) * 512 + half * 16;
    const int ldOff = r * 32 + half * 16;

    // preload k0 = 0 into buffer 0
    stage32(aSrc, &At[0][ldOff]);
    stage32(bSrc, &Bt[0][ldOff]);

    int buf = 0;
    for (int k0 = 0; k0 < 512; k0 += 32, buf ^= 1) {
        if (k0 + 32 < 512) {
            stage32(aSrc + k0 + 32, &At[buf ^ 1][ldOff]);
            stage32(bSrc + k0 + 32, &Bt[buf ^ 1][ldOff]);
            WAIT_ASYNC(4);      // the 4 older ops (current buffer) have retired
        } else {
            WAIT_ASYNC(0);
        }
        __syncthreads();

        const unsigned short* Atb = At[buf];
        const unsigned short* Btb = Bt[buf];

        v16bf bfrag[2];
#pragma unroll
        for (int ns = 0; ns < 2; ++ns) {
            int row = wn * 32 + ns * 16 + (lane & 15);
            int off = row * 32 + (lane >> 4) * 8;
            bfrag[ns] = ld16(Btb + off, Btb + off + 16);
        }
#pragma unroll
        for (int ms = 0; ms < 4; ++ms) {
            int row = wm * 64 + ms * 16 + (lane & 15);
            int off = row * 32 + (lane >> 4) * 8;
            v16bf afrag = ld16(Atb + off, Atb + off + 16);
#pragma unroll
            for (int ns = 0; ns < 2; ++ns)
                acc[ms][ns] = __builtin_amdgcn_wmma_f32_16x16x32_bf16(
                    false, afrag, false, bfrag[ns], (short)0, acc[ms][ns], false, false);
        }
        __syncthreads();   // done reading buf before next iter overwrites it
    }

    const int nbase = n0 + wn * 32 + (lane & 15);
#pragma unroll
    for (int ms = 0; ms < 4; ++ms) {
        const int mrow = m0 + wm * 64 + ms * 16 + ((lane < 16) ? 0 : 8);
#pragma unroll
        for (int ns = 0; ns < 2; ++ns) {
            const int n = nbase + ns * 16;
            const float bv = bias[n];
            if (MODE == 0) {
                unsigned short* out = (unsigned short*)outp;
#pragma unroll
                for (int v = 0; v < 8; ++v)
                    out[((size_t)b * HW + mrow + v) * N + n] = f2bf(acc[ms][ns][v] + bv);
            } else {
                float* out = (float*)outp;
#pragma unroll
                for (int v = 0; v < 8; ++v) {
                    size_t idx = ((size_t)b * N + n) * HW + (mrow + v);
                    out[idx] = acc[ms][ns][v] + bv + resid[idx];
                }
            }
        }
    }
}

// ---------------------------------------------------------------------------
// Fused flash attention: one block per (batch, head, 128-row Q tile).
// 8 waves, wave owns 16 Q rows. Loop 16 m-tiles of 64 keys:
//   S = Q K^T (WMMA), online softmax (shfl width-16 row reductions),
//   P -> bf16 via per-wave LDS, O += P V^T (WMMA).
// ---------------------------------------------------------------------------
__global__ __launch_bounds__(256) void attn_kernel(
    const unsigned short* __restrict__ qkv,  // [b][HW][1536] bf16
    unsigned short* __restrict__ out)        // [b][HW][512]  bf16
{
    const int nt = blockIdx.x, h = blockIdx.y, b = blockIdx.z;
    const int tid = threadIdx.x;
    const int lane = tid & 31, wave = tid >> 5;

    __shared__ __align__(32) unsigned short QP[128 * 64];  // Q tile, then P tiles
    __shared__ __align__(32) unsigned short Kt[64 * 64];   // [m][d]
    __shared__ __align__(32) unsigned short Vt[64 * 64];   // [d][m]

    const size_t qbase = (size_t)b * HW * 1536;

    {   // stage Q tile 128 x 64 (each thread owns 64 bytes)
        const int row = tid >> 1, hf = tid & 1;
        const unsigned short* src = qkv + qbase + (size_t)(nt * 128 + row) * 1536 + h * 64 + hf * 32;
        unsigned short* dst = QP + row * 64 + hf * 32;
        stage32(src, dst);
        stage32(src + 16, dst + 16);
    }
    WAIT_ASYNC(0);
    __syncthreads();

    v16bf qf[2];
    {
        const int row = wave * 16 + (lane & 15);
#pragma unroll
        for (int kk = 0; kk < 2; ++kk) {
            int off = row * 64 + kk * 32 + (lane >> 4) * 8;
            qf[kk] = ld16(QP + off, QP + off + 16);
        }
    }
    __syncthreads();   // all waves done with Q before QP is reused for P

    v8f oacc[4];
    float rmax[8], lsum[8];
#pragma unroll
    for (int i = 0; i < 4; ++i)
#pragma unroll
        for (int v = 0; v < 8; ++v) oacc[i][v] = 0.f;
#pragma unroll
    for (int v = 0; v < 8; ++v) { rmax[v] = -1e30f; lsum[v] = 0.f; }

    unsigned short* pw = QP + wave * (16 * 64);

    for (int ms = 0; ms < 16; ++ms) {
        const int m0 = ms * 64;
        __syncthreads();
        {   // stage K tile [m][d] (async)
            const int row = tid >> 2, q = tid & 3;
            stage32(qkv + qbase + (size_t)(m0 + row) * 1536 + 512 + h * 64 + q * 16,
                    Kt + row * 64 + q * 16);
        }
        {   // stage V transposed -> Vt[d][m] (element permute: VGPR path)
            const int m = tid >> 2, d0 = (tid & 3) * 16;
            const unsigned short* src = qkv + qbase + (size_t)(m0 + m) * 1536 + 1024 + h * 64 + d0;
            uint4 t0 = ((const uint4*)src)[0];
            uint4 t1 = ((const uint4*)src)[1];
            const unsigned short* tv = (const unsigned short*)&t0;
#pragma unroll
            for (int j = 0; j < 8; ++j) Vt[(d0 + j) * 64 + m] = tv[j];
            tv = (const unsigned short*)&t1;
#pragma unroll
            for (int j = 0; j < 8; ++j) Vt[(d0 + 8 + j) * 64 + m] = tv[j];
        }
        WAIT_ASYNC(0);
        __syncthreads();

        // S = Q K^T : 4 column subtiles x 2 k-steps
        v8f sacc[4];
#pragma unroll
        for (int i = 0; i < 4; ++i)
#pragma unroll
            for (int v = 0; v < 8; ++v) sacc[i][v] = 0.f;
#pragma unroll
        for (int nsub = 0; nsub < 4; ++nsub) {
            const int krow = nsub * 16 + (lane & 15);
#pragma unroll
            for (int kk = 0; kk < 2; ++kk) {
                int off = krow * 64 + kk * 32 + (lane >> 4) * 8;
                v16bf kf = ld16(Kt + off, Kt + off + 16);
                sacc[nsub] = __builtin_amdgcn_wmma_f32_16x16x32_bf16(
                    false, qf[kk], false, kf, (short)0, sacc[nsub], false, false);
            }
        }

        // scale + per-row max (row = vgpr index x wave-half; reduce over 16 lanes)
        float mloc[8];
#pragma unroll
        for (int v = 0; v < 8; ++v) mloc[v] = -1e30f;
#pragma unroll
        for (int nsub = 0; nsub < 4; ++nsub)
#pragma unroll
            for (int v = 0; v < 8; ++v) {
                float s = sacc[nsub][v] * 0.125f;  // 1/sqrt(64)
                sacc[nsub][v] = s;
                mloc[v] = fmaxf(mloc[v], s);
            }
#pragma unroll
        for (int v = 0; v < 8; ++v) {
            float m = mloc[v];
#pragma unroll
            for (int w = 1; w < 16; w <<= 1) m = fmaxf(m, __shfl_xor(m, w, 16));
            mloc[v] = m;
        }

        float psum[8];
#pragma unroll
        for (int v = 0; v < 8; ++v) {
            float nm = fmaxf(rmax[v], mloc[v]);
            float fac = __expf(rmax[v] - nm);
            rmax[v] = nm;
            lsum[v] *= fac;
#pragma unroll
            for (int i = 0; i < 4; ++i) oacc[i][v] *= fac;
            psum[v] = 0.f;
        }

        // P = exp(S - rmax): accumulate row sums, spill to per-wave LDS as bf16
#pragma unroll
        for (int nsub = 0; nsub < 4; ++nsub)
#pragma unroll
            for (int v = 0; v < 8; ++v) {
                float p = __expf(sacc[nsub][v] - rmax[v]);
                psum[v] += p;
                int row = (lane < 16) ? v : (v + 8);
                pw[row * 64 + nsub * 16 + (lane & 15)] = f2bf(p);
            }
#pragma unroll
        for (int v = 0; v < 8; ++v) {
            float s = psum[v];
#pragma unroll
            for (int w = 1; w < 16; w <<= 1) s += __shfl_xor(s, w, 16);
            lsum[v] += s;
        }

        // O += P V^T : A = P (16 x 64), B = Vt
#pragma unroll
        for (int kk = 0; kk < 2; ++kk) {
            int poff = (lane & 15) * 64 + kk * 32 + (lane >> 4) * 8;
            v16bf pf = ld16(pw + poff, pw + poff + 16);
#pragma unroll
            for (int dsub = 0; dsub < 4; ++dsub) {
                int voff = (dsub * 16 + (lane & 15)) * 64 + kk * 32 + (lane >> 4) * 8;
                v16bf vf = ld16(Vt + voff, Vt + voff + 16);
                oacc[dsub] = __builtin_amdgcn_wmma_f32_16x16x32_bf16(
                    false, pf, false, vf, (short)0, oacc[dsub], false, false);
            }
        }
    }

    // normalize + store [b][px][h*64+d] as bf16
    const size_t obase = ((size_t)b * HW + nt * 128 + wave * 16) * 512 + h * 64;
#pragma unroll
    for (int v = 0; v < 8; ++v) {
        float inv = 1.f / lsum[v];
        int row = (lane < 16) ? v : (v + 8);
#pragma unroll
        for (int dsub = 0; dsub < 4; ++dsub)
            out[obase + (size_t)row * 512 + dsub * 16 + (lane & 15)] = f2bf(oacc[dsub][v] * inv);
    }
}

// ---------------------------------------------------------------------------
extern "C" void kernel_launch(void* const* d_in, const int* in_sizes, int n_in,
                              void* d_out, int out_size, void* d_ws, size_t ws_size,
                              hipStream_t stream)
{
    const float* x      = (const float*)d_in[0];
    const float* nscale = (const float*)d_in[1];
    const float* nbias  = (const float*)d_in[2];
    const float* qkv_w  = (const float*)d_in[3];
    const float* qkv_b  = (const float*)d_in[4];
    const float* proj_w = (const float*)d_in[5];
    const float* proj_b = (const float*)d_in[6];
    float* out = (float*)d_out;

    char* ws = (char*)d_ws;
    unsigned short* xn    = (unsigned short*)ws; ws += (size_t)NB * HW * NC * 2;       // 32 MB
    unsigned short* wqkv  = (unsigned short*)ws; ws += (size_t)3 * NC * NC * 2;        // 1.5 MB
    unsigned short* wproj = (unsigned short*)ws; ws += (size_t)NC * NC * 2;            // 0.5 MB
    unsigned short* qkvb  = (unsigned short*)ws; ws += (size_t)NB * HW * 3 * NC * 2;   // 96 MB
    unsigned short* attnb = (unsigned short*)ws;                                       // 32 MB

    groupnorm_kernel<<<dim3(8, NB), 256, 0, stream>>>(x, nscale, nbias, xn);
    f32_to_bf16_kernel<<<(3 * NC * NC + 255) / 256, 256, 0, stream>>>(qkv_w, wqkv, 3 * NC * NC);
    f32_to_bf16_kernel<<<(NC * NC + 255) / 256, 256, 0, stream>>>(proj_w, wproj, NC * NC);
    gemm_kernel<0><<<dim3(12, 8, NB), 256, 0, stream>>>(xn, wqkv, qkv_b, nullptr, qkvb, 3 * NC);
    attn_kernel<<<dim3(8, NHEADS, NB), 256, 0, stream>>>(qkvb, attnb);
    gemm_kernel<1><<<dim3(4, 8, NB), 256, 0, stream>>>(attnb, wproj, proj_b, x, out, NC);
}
